// TFRelPartialLearnableMultiHeadAttn_35613868819139
// MI455X (gfx1250) — matrix operand order repaired and checked by reading, not compiled, not run on hardware
//
#include <hip/hip_runtime.h>
#include <hip/hip_bf16.h>

#define QLEN   1024
#define MLEN   1024
#define KLEN   2048
#define RLEN   2048
#define BSZ    4
#define NHEAD  16
#define DHEAD  64
#define DMODEL 1024

typedef __bf16 v16bf __attribute__((ext_vector_type(16)));
typedef __bf16 v8bf  __attribute__((ext_vector_type(8)));
typedef float  v8f   __attribute__((ext_vector_type(8)));

static __device__ __forceinline__ v8f wmma_bf16(v16bf a, v16bf b, v8f c) {
  // D = A(16x32 bf16) * B(32x16 bf16) + C(16x16 f32)
  return __builtin_amdgcn_wmma_f32_16x16x32_bf16(false, a, false, b, (short)0, c, false, false);
}

// ---------------------------------------------------------------------------
// Async global->LDS staging (gfx1250 GLOBAL_LOAD_ASYNC_TO_LDS_B128, ASYNCcnt)
// with a synchronous fallback if the builtin is unavailable.
// Probe-confirmed signature: (int4 AS1*, int4 AS3*, imm int offset, imm int cpol)
// ---------------------------------------------------------------------------
#if defined(__AMDGCN__) && __has_builtin(__builtin_amdgcn_global_load_async_to_lds_b128)
#define USE_ASYNC_LDS 1
#endif

typedef int v4i __attribute__((vector_size(16)));
typedef __attribute__((address_space(1))) v4i as1_v4i;
typedef __attribute__((address_space(3))) v4i as3_v4i;

static __device__ __forceinline__ void stage_b128(const __bf16* gsrc, __bf16* ldst) {
#ifdef USE_ASYNC_LDS
  __builtin_amdgcn_global_load_async_to_lds_b128(
      (as1_v4i*)gsrc, (as3_v4i*)ldst, 0, 0);
#else
  *(uint4*)ldst = *(const uint4*)gsrc;
#endif
}

static __device__ __forceinline__ void wait_async_stage() {
#ifdef USE_ASYNC_LDS
#if __has_builtin(__builtin_amdgcn_s_wait_asynccnt)
  __builtin_amdgcn_s_wait_asynccnt(0);
#else
  asm volatile("s_wait_asynccnt 0" ::: "memory");
#endif
#endif
}

// ---------------------------------------------------------------------------
// Kernel 0a: f32 -> bf16 flat convert (grid-stride)
// ---------------------------------------------------------------------------
__global__ __launch_bounds__(256) void conv_bf16_kernel(
    const float* __restrict__ src, __bf16* __restrict__ dst, int n) {
  for (int i = blockIdx.x * 256 + threadIdx.x; i < n; i += gridDim.x * 256)
    dst[i] = (__bf16)src[i];
}

// ---------------------------------------------------------------------------
// Kernel 0b: W [K][N] f32 -> Wt [N][K] bf16 (tiled transpose through LDS)
// Block 256 (32x8), tile 32x32. grid = (N/32, K/32)
// ---------------------------------------------------------------------------
__global__ __launch_bounds__(256) void transpose_w_kernel(
    const float* __restrict__ W, __bf16* __restrict__ Wt, int K, int N) {
  __shared__ float tile[32][33];
  const int k0 = blockIdx.y * 32, n0 = blockIdx.x * 32;
  const int r = threadIdx.x >> 5;   // 0..7
  const int c = threadIdx.x & 31;   // 0..31
#pragma unroll
  for (int i = 0; i < 32; i += 8)
    tile[r + i][c] = W[(size_t)(k0 + r + i) * N + n0 + c];
  __syncthreads();
#pragma unroll
  for (int i = 0; i < 32; i += 8)
    Wt[(size_t)(n0 + r + i) * K + k0 + c] = (__bf16)tile[c][r + i];
}

// ---------------------------------------------------------------------------
// Per-wave 32x64 GEMM strip, all-bf16 operands:
//   A  : bf16 [M][Ktot] row-major (k fastest)
//   Bt : bf16 [N][Ktot] (pre-transposed, k fastest)
// Two 16-row A fragments share the four B fragments -> 8 WMMAs per 8
// ds_load_b128 (1:1). B tile (64n x 32k, 4KB) staged via async b128 copies,
// stride-40 rows (16B aligned, conflict-free).
// Block = 256 threads (8 waves); each wave owns rows [mrow0, mrow0+32).
// ---------------------------------------------------------------------------
static __device__ __forceinline__ void gemm_strip_bf(
    const __bf16* __restrict__ A, int lda,
    const __bf16* __restrict__ Bt, int ldb,
    int K, int mrow0, int n0,
    __bf16* sB /* 64*40 */, v8f acc[2][4]) {
  const int lane = threadIdx.x & 31;
  const int hl = lane >> 4, mr = lane & 15;
  const int tn = threadIdx.x >> 2;  // 0..63  : n within tile
  const int tq = threadIdx.x & 3;   // 0..3   : 8-elem chunk within k row
  const __bf16* arow0 = A + (size_t)(mrow0 + mr) * lda;
  const __bf16* arow1 = A + (size_t)(mrow0 + 16 + mr) * lda;
  const __bf16* brow  = Bt + (size_t)(n0 + tn) * ldb + tq * 8;

  for (int k0 = 0; k0 < K; k0 += 32) {
    // issue A-fragment loads early: latency overlaps barrier + staging
    v8bf a0lo = *(const v8bf*)(arow0 + k0 + hl * 16);
    v8bf a0hi = *(const v8bf*)(arow0 + k0 + hl * 16 + 8);
    v8bf a1lo = *(const v8bf*)(arow1 + k0 + hl * 16);
    v8bf a1hi = *(const v8bf*)(arow1 + k0 + hl * 16 + 8);

    __syncthreads();
    // stage Bt[n0..n0+63][k0..k0+31] -> LDS (16B per thread, async)
    stage_b128(brow + k0, sB + tn * 40 + tq * 8);
    if (k0 + 32 < K)
      __builtin_prefetch(brow + k0 + 32, 0, 3);
    wait_async_stage();
    __syncthreads();

    v16bf af0 = __builtin_shufflevector(a0lo, a0hi, 0, 1, 2, 3, 4, 5, 6, 7,
                                        8, 9, 10, 11, 12, 13, 14, 15);
    v16bf af1 = __builtin_shufflevector(a1lo, a1hi, 0, 1, 2, 3, 4, 5, 6, 7,
                                        8, 9, 10, 11, 12, 13, 14, 15);
    // hoist all B fragments (8x ds_load_b128); each feeds two WMMAs
    v16bf bfr[4];
#pragma unroll
    for (int t = 0; t < 4; ++t)
      bfr[t] = *(const v16bf*)(sB + (t * 16 + mr) * 40 + hl * 16);
#pragma unroll
    for (int t = 0; t < 4; ++t) {
      acc[0][t] = wmma_bf16(af0, bfr[t], acc[0][t]);
      acc[1][t] = wmma_bf16(af1, bfr[t], acc[1][t]);
    }
  }
}

// ---------------------------------------------------------------------------
// Kernel 1: w_heads = cat_bf16 @ Wqkv_t  (8192 x 1024 @ -> 3072 cols)
// Scatter epilogue: Q -> f32 [b][h][q][d], K -> bf16 [b][h][k][d],
//                   V -> bf16 transposed [b][h][d][k]
// ---------------------------------------------------------------------------
__global__ __launch_bounds__(256) void qkv_gemm_kernel(
    const __bf16* __restrict__ Abf, const __bf16* __restrict__ Wqkvt,
    float* __restrict__ Qf, __bf16* __restrict__ Kb, __bf16* __restrict__ Vt) {
  __shared__ __bf16 sB[64 * 40];
  const int wave = threadIdx.x >> 5;
  const int m0 = blockIdx.y * 256 + wave * 32;
  const int n0 = blockIdx.x * 64;

  v8f acc[2][4] = {};
  gemm_strip_bf(Abf, DMODEL, Wqkvt, DMODEL, DMODEL, m0, n0, sB, acc);

  const int lane = threadIdx.x & 31, hl = lane >> 4, mr = lane & 15;
#pragma unroll
  for (int rg = 0; rg < 2; ++rg)
#pragma unroll
    for (int t = 0; t < 4; ++t)
#pragma unroll
      for (int v = 0; v < 8; ++v) {
        int mg = m0 + rg * 16 + v + 8 * hl;  // row in cat: kpos*BSZ + b
        int c  = n0 + t * 16 + mr;           // column in [0, 3072)
        float val = acc[rg][t][v];
        int kpos = mg >> 2, b = mg & 3;
        int sect = c >> 10, hd = c & 1023;
        int hh = hd >> 6, dd = hd & 63;
        if (sect == 0) {
          if (kpos >= MLEN)
            Qf[(((size_t)b * NHEAD + hh) * QLEN + (kpos - MLEN)) * DHEAD + dd] = val;
        } else if (sect == 1) {
          Kb[(((size_t)b * NHEAD + hh) * KLEN + kpos) * DHEAD + dd] = (__bf16)val;
        } else {
          Vt[(((size_t)b * NHEAD + hh) * DHEAD + dd) * KLEN + kpos] = (__bf16)val;
        }
      }
}

// ---------------------------------------------------------------------------
// Kernel 2: r_head_k = r_bf16 @ Wr_t -> bf16 [h][p][d]
// ---------------------------------------------------------------------------
__global__ __launch_bounds__(256) void r_gemm_kernel(
    const __bf16* __restrict__ Rbf, const __bf16* __restrict__ Wrt,
    __bf16* __restrict__ Rb) {
  __shared__ __bf16 sB[64 * 40];
  const int wave = threadIdx.x >> 5;
  const int m0 = blockIdx.y * 256 + wave * 32;
  const int n0 = blockIdx.x * 64;

  v8f acc[2][4] = {};
  gemm_strip_bf(Rbf, DMODEL, Wrt, DMODEL, DMODEL, m0, n0, sB, acc);

  const int lane = threadIdx.x & 31, hl = lane >> 4, mr = lane & 15;
#pragma unroll
  for (int rg = 0; rg < 2; ++rg)
#pragma unroll
    for (int t = 0; t < 4; ++t)
#pragma unroll
      for (int v = 0; v < 8; ++v) {
        int p = m0 + rg * 16 + v + 8 * hl;
        int c = n0 + t * 16 + mr;
        int hh = c >> 6, dd = c & 63;
        Rb[((size_t)hh * RLEN + p) * DHEAD + dd] = (__bf16)acc[rg][t][v];
      }
}

// ---------------------------------------------------------------------------
// Kernel 3: fused rel-attention, flash style. 1 wave per (b, h, 16-query tile).
// score[i,j] = (q_i+rwb)·k_j + (q_i+rrb)·r_{j+QLEN-1-i}, masked j<=i+MLEN.
// Per 32-key tile: 4 WMMA (AC) + 6 WMMA (BD band, 48 cols) + 4 WMMA (PV).
// All fragments hoisted before each WMMA group; sP stored in A-fragment
// contiguous (k = c) order so the P fragment is one contiguous 32B read.
// ---------------------------------------------------------------------------
__global__ __launch_bounds__(32) void attn_kernel(
    const float* __restrict__ Qf, const __bf16* __restrict__ Kb,
    const __bf16* __restrict__ Vt, const __bf16* __restrict__ Rb,
    const float* __restrict__ rwb, const float* __restrict__ rrb,
    __bf16* __restrict__ AVbf) {
  const int qt = blockIdx.x, h = blockIdx.y, b = blockIdx.z;
  const int i0 = qt * 16;
  const int lane = threadIdx.x & 31, hl = lane >> 4, mr = lane & 15;

  __shared__ float  sS[16][32];
  __shared__ float  sBD[16][48];
  __shared__ float  sFac[16];
  __shared__ float  sRowM[16];
  __shared__ float  sRowL[16];
  __shared__ __bf16 sP[16][48];   // stride 48 (96B) keeps 16B alignment

  // Q row for this lane; build A fragments for (q + r_w_bias), (q + r_r_bias)
  const float* qrow = Qf + (((size_t)b * NHEAD + h) * QLEN + (i0 + mr)) * DHEAD;
  const float* bw = rwb + h * DHEAD;
  const float* br = rrb + h * DHEAD;
  v16bf aw[2], ar[2];
#pragma unroll
  for (int kh = 0; kh < 2; ++kh)
#pragma unroll
    for (int v = 0; v < 8; ++v) {
      int kk = kh * 32 + hl * 16 + 2 * v;
      float x0 = qrow[kk], x1 = qrow[kk + 1];
      aw[kh][2 * v]     = (__bf16)(x0 + bw[kk]);
      aw[kh][2 * v + 1] = (__bf16)(x1 + bw[kk + 1]);
      ar[kh][2 * v]     = (__bf16)(x0 + br[kk]);
      ar[kh][2 * v + 1] = (__bf16)(x1 + br[kk + 1]);
    }

  if (lane < 16) { sRowM[lane] = -3.0e38f; sRowL[lane] = 0.f; }
  v8f o[4] = {};

  const __bf16* Kbase = Kb + ((size_t)b * NHEAD + h) * KLEN * DHEAD;
  const __bf16* Vbase = Vt + ((size_t)b * NHEAD + h) * DHEAD * KLEN;
  const __bf16* Rbase = Rb + (size_t)h * RLEN * DHEAD;
  const float scale = 0.125f;  // 1/sqrt(64)
  const int jlast  = i0 + 15 + MLEN;
  const int ntiles = (jlast + 32) / 32;

  for (int t = 0; t < ntiles; ++t) {
    const int j0 = t * 32;
    __builtin_prefetch(Kbase + (size_t)(j0 + 32 + mr) * DHEAD, 0, 3);

    // ---- AC: hoist 4 K fragments, then 4 WMMAs ----
    v16bf kf[4];
#pragma unroll
    for (int st = 0; st < 2; ++st)
#pragma unroll
      for (int kh = 0; kh < 2; ++kh)
        kf[st * 2 + kh] = *(const v16bf*)(Kbase +
            (size_t)(j0 + st * 16 + mr) * DHEAD + kh * 32 + hl * 16);
    v8f accS[2] = {};
    accS[0] = wmma_bf16(aw[0], kf[0], accS[0]);
    accS[0] = wmma_bf16(aw[1], kf[1], accS[0]);
    accS[1] = wmma_bf16(aw[0], kf[2], accS[1]);
    accS[1] = wmma_bf16(aw[1], kf[3], accS[1]);

    // ---- BD band: hoist 6 R fragments, then 6 WMMAs ----
    const int kbase = j0 + QLEN - 16 - i0;  // rk idx base (j0+QLEN-1-i0-15)
    v16bf rf[6];
#pragma unroll
    for (int st = 0; st < 3; ++st) {
      int p = kbase + st * 16 + mr;
#pragma unroll
      for (int kh = 0; kh < 2; ++kh) {
        v16bf bb = {};
        if ((unsigned)p < (unsigned)RLEN)
          bb = *(const v16bf*)(Rbase + (size_t)p * DHEAD + kh * 32 + hl * 16);
        rf[st * 2 + kh] = bb;
      }
    }
    v8f accB[3] = {};
#pragma unroll
    for (int st = 0; st < 3; ++st) {
      accB[st] = wmma_bf16(ar[0], rf[st * 2 + 0], accB[st]);
      accB[st] = wmma_bf16(ar[1], rf[st * 2 + 1], accB[st]);
    }

    // dump tiles to LDS
#pragma unroll
    for (int v = 0; v < 8; ++v) {
      int rr = v + 8 * hl;
      sS[rr][mr]       = accS[0][v];
      sS[rr][16 + mr]  = accS[1][v];
      sBD[rr][mr]      = accB[0][v];
      sBD[rr][16 + mr] = accB[1][v];
      sBD[rr][32 + mr] = accB[2][v];
    }
    __syncthreads();

    // online softmax per query row (lanes 0..15 each own one row)
    if (lane < 16) {
      const int rr = lane;
      const int iq = i0 + rr;
      float mprev = sRowM[rr];
      float mnew = mprev;
#pragma unroll
      for (int c = 0; c < 32; ++c) {
        int j = j0 + c;
        if (j <= iq + MLEN && j < KLEN) {
          float s = (sS[rr][c] + sBD[rr][c + 15 - rr]) * scale;
          mnew = fmaxf(mnew, s);
        }
      }
      float corr = __expf(mprev - mnew);
      float l = sRowL[rr] * corr;
#pragma unroll
      for (int c = 0; c < 32; ++c) {
        int j = j0 + c;
        float p = 0.f;
        if (j <= iq + MLEN && j < KLEN) {
          float s = (sS[rr][c] + sBD[rr][c + 15 - rr]) * scale;
          p = __expf(s - mnew);
        }
        // A-fragment position: contiguous (k = c)
        int pos = c;
        sP[rr][pos] = (__bf16)p;
        l += p;
      }
      sRowM[rr] = mnew;
      sRowL[rr] = l;
      sFac[rr]  = corr;
    }
    __syncthreads();

    // rescale running O, then P x V via WMMA (fragments hoisted)
#pragma unroll
    for (int v = 0; v < 8; ++v) {
      float f = sFac[v + 8 * hl];
      o[0][v] *= f; o[1][v] *= f; o[2][v] *= f; o[3][v] *= f;
    }

    v16bf pf = *(const v16bf*)(&sP[mr][hl * 16]);  // one 32B LDS read
    v16bf vf[4];
#pragma unroll
    for (int seg = 0; seg < 4; ++seg)
      vf[seg] = *(const v16bf*)(Vbase + (size_t)(seg * 16 + mr) * KLEN +
                                j0 + hl * 16);
#pragma unroll
    for (int seg = 0; seg < 4; ++seg)
      o[seg] = wmma_bf16(pf, vf[seg], o[seg]);
    __syncthreads();  // protect sS/sBD/sP before next iteration
  }

  // normalize and store attn_vec bf16 [q][b][h*64+d]
#pragma unroll
  for (int seg = 0; seg < 4; ++seg)
#pragma unroll
    for (int v = 0; v < 8; ++v) {
      int rr = v + 8 * hl;
      float invl = 1.0f / sRowL[rr];
      int iq = i0 + rr;
      AVbf[(((size_t)iq * BSZ + b) * NHEAD + h) * DHEAD + seg * 16 + mr] =
          (__bf16)(o[seg][v] * invl);
    }
}

// ---------------------------------------------------------------------------
// Kernel 4: attn_out = attn_vec @ W_o  (4096 x 1024 @ -> 1024 cols)
// ---------------------------------------------------------------------------
__global__ __launch_bounds__(256) void out_gemm_kernel(
    const __bf16* __restrict__ AVbf, const __bf16* __restrict__ Wot,
    float* __restrict__ OW) {
  __shared__ __bf16 sB[64 * 40];
  const int wave = threadIdx.x >> 5;
  const int m0 = blockIdx.y * 256 + wave * 32;
  const int n0 = blockIdx.x * 64;

  v8f acc[2][4] = {};
  gemm_strip_bf(AVbf, DMODEL, Wot, DMODEL, DMODEL, m0, n0, sB, acc);

  const int lane = threadIdx.x & 31, hl = lane >> 4, mr = lane & 15;
#pragma unroll
  for (int rg = 0; rg < 2; ++rg)
#pragma unroll
    for (int t = 0; t < 4; ++t)
#pragma unroll
      for (int v = 0; v < 8; ++v) {
        int mg = m0 + rg * 16 + v + 8 * hl;
        int c  = n0 + t * 16 + mr;
        OW[(size_t)mg * DMODEL + c] = acc[rg][t][v];
      }
}

// ---------------------------------------------------------------------------
// Kernel 5: out = layer_norm(w + attn_out)   (one 256-thread block per row)
// ---------------------------------------------------------------------------
__global__ __launch_bounds__(256) void ln_kernel(
    const float* __restrict__ w, const float* __restrict__ OW,
    const float* __restrict__ gamma, const float* __restrict__ beta,
    float* __restrict__ out) {
  const int m = blockIdx.x;
  const int tid = threadIdx.x;
  __shared__ float red[256];

  const float* a = w + (size_t)m * DMODEL;
  const float* o = OW + (size_t)m * DMODEL;
  float vals[4];
  float s = 0.f;
#pragma unroll
  for (int k = 0; k < 4; ++k) {
    vals[k] = a[tid + k * 256] + o[tid + k * 256];
    s += vals[k];
  }
  red[tid] = s;
  __syncthreads();
  for (int off = 128; off > 0; off >>= 1) {
    if (tid < off) red[tid] += red[tid + off];
    __syncthreads();
  }
  float mean = red[0] * (1.0f / DMODEL);
  __syncthreads();
  float s2 = 0.f;
#pragma unroll
  for (int k = 0; k < 4; ++k) {
    float d = vals[k] - mean;
    s2 += d * d;
  }
  red[tid] = s2;
  __syncthreads();
  for (int off = 128; off > 0; off >>= 1) {
    if (tid < off) red[tid] += red[tid + off];
    __syncthreads();
  }
  float inv = rsqrtf(red[0] * (1.0f / DMODEL) + 1e-5f);
#pragma unroll
  for (int k = 0; k < 4; ++k)
    out[(size_t)m * DMODEL + tid + k * 256] =
        (vals[k] - mean) * inv * gamma[tid + k * 256] + beta[tid + k * 256];
}

// ---------------------------------------------------------------------------
extern "C" void kernel_launch(void* const* d_in, const int* in_sizes, int n_in,
                              void* d_out, int out_size, void* d_ws, size_t ws_size,
                              hipStream_t stream) {
  const float* w     = (const float*)d_in[0];
  const float* r     = (const float*)d_in[1];
  const float* mems  = (const float*)d_in[2];
  const float* Wqkv  = (const float*)d_in[3];
  const float* Wr    = (const float*)d_in[4];
  const float* Wo    = (const float*)d_in[5];
  const float* gamma = (const float*)d_in[6];
  const float* beta  = (const float*)d_in[7];
  const float* rwb   = (const float*)d_in[8];
  const float* rrb   = (const float*)d_in[9];
  // d_in[10] = attn_mask: computed analytically (j <= i + MLEN), not read.
  float* out = (float*)d_out;

  char* ws = (char*)d_ws;
  const size_t MB = 1u << 20;
  __bf16* Abf   = (__bf16*)(ws);              // 16 MB bf16 [8192][1024] (cat)
  __bf16* Rbf   = (__bf16*)(ws + 16 * MB);    //  4 MB bf16 [2048][1024]
  __bf16* Wqkvt = (__bf16*)(ws + 20 * MB);    //  6 MB bf16 [3072][1024]
  __bf16* Wrt   = (__bf16*)(ws + 26 * MB);    //  2 MB bf16 [1024][1024]
  __bf16* Wot   = (__bf16*)(ws + 28 * MB);    //  2 MB bf16 [1024][1024]
  float*  Qf    = (float*) (ws + 30 * MB);    // 16 MB f32  [b][h][q][d]
  __bf16* Kb    = (__bf16*)(ws + 46 * MB);    // 16 MB bf16 [b][h][k][d]
  __bf16* Vt    = (__bf16*)(ws + 62 * MB);    // 16 MB bf16 [b][h][d][k]
  __bf16* Rb    = (__bf16*)(ws + 78 * MB);    //  4 MB bf16 [h][p][d]
  __bf16* AVbf  = (__bf16*)(ws + 82 * MB);    //  8 MB bf16 [4096][1024]
  float*  OW    = (float*) (ws + 90 * MB);    // 16 MB f32  [4096][1024]

  // 0) convert activations, transpose+convert weights
  conv_bf16_kernel<<<1024, 256, 0, stream>>>(mems, Abf, MLEN * BSZ * DMODEL);
  conv_bf16_kernel<<<1024, 256, 0, stream>>>(w, Abf + (size_t)MLEN * BSZ * DMODEL,
                                             QLEN * BSZ * DMODEL);
  conv_bf16_kernel<<<1024, 256, 0, stream>>>(r, Rbf, RLEN * DMODEL);
  transpose_w_kernel<<<dim3(96, 32), 256, 0, stream>>>(Wqkv, Wqkvt, DMODEL, 3072);
  transpose_w_kernel<<<dim3(32, 32), 256, 0, stream>>>(Wr, Wrt, DMODEL, DMODEL);
  transpose_w_kernel<<<dim3(32, 32), 256, 0, stream>>>(Wo, Wot, DMODEL, DMODEL);

  // 1..5) projections, fused attention, output projection, layernorm
  qkv_gemm_kernel<<<dim3(48, 32), 256, 0, stream>>>(Abf, Wqkvt, Qf, Kb, Vt);
  r_gemm_kernel<<<dim3(16, 8), 256, 0, stream>>>(Rbf, Wrt, Rb);
  attn_kernel<<<dim3(64, 16, 4), 32, 0, stream>>>(Qf, Kb, Vt, Rb, rwb, rrb, AVbf);
  out_gemm_kernel<<<dim3(16, 16), 256, 0, stream>>>(AVbf, Wot, OW);
  ln_kernel<<<dim3(4096), 256, 0, stream>>>(w, OW, gamma, beta, out);
}
